// LorentzSelfAttention_46256797777933
// MI455X (gfx1250) — matrix operand order repaired and checked by reading, not compiled, not run on hardware
//
#include <hip/hip_runtime.h>
#include <hip/hip_bf16.h>

// Problem constants (reference: B=8, L=2048, D=128, K_CURV=1)
#define BB 8
#define LL 2048
#define DD 128

typedef __attribute__((ext_vector_type(16))) _Float16 v16h;
typedef __attribute__((ext_vector_type(8)))  _Float16 v8h;
typedef __attribute__((ext_vector_type(8)))  float    v8f;

// Tensor Data Mover availability (ROCm7.2 clang-22: 5-arg; upstream clang-23: 6-arg)
#if defined(__has_builtin)
#if __has_builtin(__builtin_amdgcn_tensor_load_to_lds) && __has_builtin(__builtin_amdgcn_s_wait_tensorcnt)
#define HAVE_TDM 1
#endif
#endif

#ifdef HAVE_TDM
typedef __attribute__((ext_vector_type(4))) unsigned int u32x4;
typedef __attribute__((ext_vector_type(4))) int          i32x4;
typedef __attribute__((ext_vector_type(8))) int          i32x8;
#endif

static __device__ __forceinline__ float neg_inf_val() { return -4294967295.0f; } // -(2^32-1)

// A-operand K index for V_WMMA_F32_16X16X32_F16 (16-bit A 16x32 layout, ISA 7.12.2):
// lanes 0-15: kb=0, lanes 16-31: kb=8; VGPR j<4 -> K=kb+2j, j>=4 -> K=16+kb+2(j-4)
static __device__ __forceinline__ int wmma_a_k(int j, int lane) {
  int kb = (lane >> 4) * 8;
  return (j < 4) ? (kb + 2 * j) : (16 + kb + 2 * (j - 4));
}

// Reductions across the 16-lane half-wave (xor masks 1,2,4,8 never cross bit 4).
static __device__ __forceinline__ float red16_max(float v) {
  v = fmaxf(v, __shfl_xor(v, 1, 32));
  v = fmaxf(v, __shfl_xor(v, 2, 32));
  v = fmaxf(v, __shfl_xor(v, 4, 32));
  v = fmaxf(v, __shfl_xor(v, 8, 32));
  return v;
}
static __device__ __forceinline__ float red16_sum(float v) {
  v += __shfl_xor(v, 1, 32);
  v += __shfl_xor(v, 2, 32);
  v += __shfl_xor(v, 4, 32);
  v += __shfl_xor(v, 8, 32);
  return v;
}

// ---------------------------------------------------------------------------
// Kernel 1: Lorentz linear projection.
//   out_row = x @ W^T + b  (16x128 tile per block, 8 waves x one 16x16 WMMA tile)
//   then time = sigmoid(c0)*exp(scale)+1.1 ; s=(t^2-1)/sum(c[1:]^2);
//   emit f16 [time (optionally negated), c[1:]*sqrt(s)]
// ---------------------------------------------------------------------------
__global__ __launch_bounds__(256)
void lorentz_proj_f16_wmma(const float* __restrict__ x,      // [B*L, D]
                           const float* __restrict__ W,      // [D, D] row-major
                           const float* __restrict__ bias,   // [D]
                           const float* __restrict__ scale,  // [1]
                           _Float16* __restrict__ outh,      // [B*L, D] f16
                           int neg_time)
{
  __shared__ float T[16][DD + 1];
  __shared__ float part[16][16];
  __shared__ float time_s[16], sfac_s[16];

  const int tid  = threadIdx.x;
  const int wave = tid >> 5;
  const int lane = tid & 31;
  const int hf   = lane >> 4;    // which half of the wave
  const int m16  = lane & 15;    // A row / C-N / B col index
  const int rowbase = blockIdx.x * 16;   // row tile in [0, B*L)
  const int colbase = wave * 16;         // this wave's 16 output columns

  const float* xrow = x + (size_t)(rowbase + m16) * DD;
  const float* wrow = W + (size_t)(colbase + m16) * DD; // out[m][n]=sum_k x[m][k]*W[n][k]

  v8f c = {};
#pragma unroll
  for (int ks = 0; ks < 4; ++ks) {            // K = 128 = 4 * 32
    v16h a, bmat;
#pragma unroll
    for (int j = 0; j < 8; ++j) {
      int k0 = ks * 32 + wmma_a_k(j, lane);
      a[2 * j]     = (_Float16)xrow[k0];
      a[2 * j + 1] = (_Float16)xrow[k0 + 1];
    }
#pragma unroll
    for (int i = 0; i < 16; ++i) {            // B 32x16: lane half selects K range
      int k = ks * 32 + hf * 8 + (i & 7) + (i >> 3) * 16;
      bmat[i] = (_Float16)wrow[k];
    }
    c = __builtin_amdgcn_wmma_f32_16x16x32_f16(false, a, false, bmat,
                                               (short)0, c, false, false);
  }

  // C layout: VGPR r -> row (r + 8*hf), col (lane&15). Stage tile in LDS.
#pragma unroll
  for (int r = 0; r < 8; ++r)
    T[r + 8 * hf][colbase + m16] = c[r] + bias[colbase + m16];
  __syncthreads();

  // Parallel row-wise sum of squares over cols [1,128): 16 threads per row.
  {
    int pr = tid >> 4, pc = tid & 15;
    float ss = 0.0f;
#pragma unroll
    for (int j = 0; j < 8; ++j) {
      int col = pc * 8 + j;
      if (col >= 1) { float v = T[pr][col]; ss += v * v; }
    }
    part[pr][pc] = ss;
  }
  __syncthreads();
  if (tid < 16) {
    float ss = 0.0f;
#pragma unroll
    for (int w = 0; w < 16; ++w) ss += part[tid][w];
    float raw0 = T[tid][0];
    float t = 1.0f / (1.0f + __expf(-raw0)) * __expf(scale[0]) + 1.1f;
    time_s[tid] = t;
    sfac_s[tid] = sqrtf((t * t - 1.0f) / ss);   // K_CURV = 1
  }
  __syncthreads();

  // Vectorized f16 output: one 16B chunk (8 halves) per thread.
  {
    int r = tid >> 4, c0 = (tid & 15) * 8;
    v8h chunk;
#pragma unroll
    for (int j = 0; j < 8; ++j) {
      int d = c0 + j;
      float val = (d == 0) ? (neg_time ? -time_s[r] : time_s[r])
                           : T[r][d] * sfac_s[r];
      chunk[j] = (_Float16)val;
    }
    *(v8h*)&outh[(size_t)(rowbase + r) * DD + c0] = chunk;
  }
}

// ---------------------------------------------------------------------------
// Kernel 2: flash attention over f16 Q (time pre-negated), K, V.
//   Per block: 16 query rows, 8 waves; key stripes of 128 up to causal diagonal.
//   S stays in registers; softmax via lane shuffles + per-row wave combine;
//   V stripe staged in LDS by the Tensor Data Mover (padded rows) overlapping
//   with the QK^T WMMAs; f32 accumulators; Lorentz normalize at the end.
// ---------------------------------------------------------------------------
__global__ __launch_bounds__(256)
void lorentz_flash_attn_wmma(const _Float16* __restrict__ Qh,
                             const _Float16* __restrict__ Kh,
                             const _Float16* __restrict__ Vh,
                             const unsigned char* __restrict__ mask, // [B, L]
                             const float* __restrict__ attn_scale,   // [1]
                             const float* __restrict__ attn_bias,    // [1]
                             float* __restrict__ out)                // [B, L, D]
{
  __shared__ __align__(16) _Float16 Qs[16][DD];
  __shared__ __align__(16) _Float16 Ps[16][DD];
  __shared__ __align__(16) _Float16 Vs[128][DD + 8]; // 272B row stride: 16B aligned, bank-spread
  __shared__ float Os[16][DD + 1];
  __shared__ float pmax[16][8], psum[16][8];
  __shared__ float rowm[16], rowl[16], rowscale[16];

  const int tid  = threadIdx.x;
  const int wave = tid >> 5;
  const int lane = tid & 31;
  const int hf   = lane >> 4;
  const int m16  = lane & 15;

  const int bidx  = blockIdx.x >> 7;        // L/16 = 128 query tiles per batch
  const int qt    = blockIdx.x & 127;
  const int qbase = qt * 16;
  const unsigned char* maskp = mask + (size_t)bidx * LL;

  // Stage the 16x128 Q tile (f16) in LDS: 256 x 16B, one uint4 per thread.
  {
    const uint4* qsrc = (const uint4*)(Qh + ((size_t)bidx * LL + qbase) * DD);
    ((uint4*)&Qs[0][0])[tid] = qsrc[tid];
  }
  if (tid < 16) { rowm[tid] = -3.0e38f; rowl[tid] = 0.0f; }
  __syncthreads();

  const float inv_scale = 1.0f / attn_scale[0];
  const float abias     = attn_bias[0];

  v8f o = {};                               // O columns [wave*16, wave*16+16)
  const int nstripes = qt / 8 + 1;          // causal: stripes covering keys <= qbase+15

  for (int kt = 0; kt < nstripes; ++kt) {
    const int kstart = kt * 128;
    const int keyw   = kstart + wave * 16 + m16;   // this lane's key column in C

    // ---- stage V stripe (128x128 f16) into LDS (padded to 136-half rows) ----
#ifdef HAVE_TDM
    if (wave == 0) {
      // Tensor DMA descriptor (D#), 2D tile: 128 rows x 128 f16, dim0 stride 128,
      // LDS padding: +4 DWORDs (16B) after every 64 DWORDs (one 256B row).
      unsigned long long ga =
          (unsigned long long)(uintptr_t)(Vh + ((size_t)bidx * LL + kstart) * DD);
      unsigned lds_off = (unsigned)(uintptr_t)&Vs[0][0];
      u32x4 g0;
      g0[0] = 1u;                                          // count=1 (valid user D#)
      g0[1] = lds_off;                                     // lds_addr
      g0[2] = (unsigned)(ga & 0xFFFFFFFFu);                // global_addr[31:0]
      g0[3] = (unsigned)((ga >> 32) & 0x1FFFFFFu) | (2u << 30); // addr[56:32] | type=2
      i32x8 g1;
      g1[0] = (int)((1u << 16)        // data_size = 2B
                  | (1u << 20)        // pad_enable
                  | (5u << 22)        // pad_interval: 64 DWORDs
                  | (3u << 25));      // pad_amount: 4 DWORDs
      g1[1] = (int)(128u << 16);      // tensor_dim0[15:0] = 128 (atomic_addr = 0)
      g1[2] = (int)(128u << 16);      // tensor_dim1[15:0] = 128
      g1[3] = (int)(128u << 16);      // tile_dim0 = 128
      g1[4] = (int)(128u);            // tile_dim1 = 128, tile_dim2 = 0
      g1[5] = (int)(128u);            // tensor_dim0_stride = 128
      g1[6] = 0;
      g1[7] = 0;
      i32x4 gz = {0, 0, 0, 0};
#if __clang_major__ >= 23
      i32x8 gz8 = {0, 0, 0, 0, 0, 0, 0, 0};
      __builtin_amdgcn_tensor_load_to_lds(g0, g1, gz, gz, gz8, 0);
#else
      __builtin_amdgcn_tensor_load_to_lds(g0, g1, gz, gz, 0);
#endif
    }
#else
    {
      const uint4* vsrc = (const uint4*)(Vh + ((size_t)bidx * LL + kstart) * DD);
      for (int i = tid; i < 128 * (DD / 8); i += 256) {   // 2048 x 16B
        int r = i >> 4, cc = i & 15;
        *(uint4*)&Vs[r][cc * 8] = vsrc[i];
      }
    }
#endif

    // Prefetch next stripe's K rows (gfx1250 global_prefetch_b8).
    if (kt + 1 < nstripes) {
      __builtin_prefetch(Kh + ((size_t)bidx * LL + kstart + 128 + (tid >> 1)) * DD, 0, 1);
    }

    // ---- S = Q K^T for this wave's 16 keys (overlaps the V TDM transfer) ----
    v8f c = {};
    const _Float16* krow = Kh + ((size_t)bidx * LL + keyw) * DD;
#pragma unroll
    for (int ks = 0; ks < 4; ++ks) {
      v16h a, bmat;
#pragma unroll
      for (int j = 0; j < 8; ++j) {
        int k0 = ks * 32 + wmma_a_k(j, lane);
        a[2 * j]     = Qs[m16][k0];
        a[2 * j + 1] = Qs[m16][k0 + 1];
      }
#pragma unroll
      for (int i = 0; i < 16; ++i)
        bmat[i] = krow[ks * 32 + hf * 8 + (i & 7) + (i >> 3) * 16];
      c = __builtin_amdgcn_wmma_f32_16x16x32_f16(false, a, false, bmat,
                                                 (short)0, c, false, false);
    }

    // Masked scores + per-row (16-lane) max; combine across waves in LDS.
    float sreg[8];
    const bool kmasked = maskp[keyw] != 0;
#pragma unroll
    for (int r = 0; r < 8; ++r) {
      int qrow = qbase + r + 8 * hf;
      float sval = (2.0f + 2.0f * c[r]) * inv_scale + abias;
      if (keyw > qrow || kmasked) sval = neg_inf_val();
      sreg[r] = sval;
      float mx = red16_max(sval);
      if (m16 == 0) pmax[r + 8 * hf][wave] = mx;
    }
#ifdef HAVE_TDM
    if (wave == 0) __builtin_amdgcn_s_wait_tensorcnt(0);  // V stripe landed in LDS
#endif
    __syncthreads();   // pmax ready; Vs staging complete (TDM waited above)

    if (tid < 16) {
      float mnew = rowm[tid];
#pragma unroll
      for (int w = 0; w < 8; ++w) mnew = fmaxf(mnew, pmax[tid][w]);
      rowscale[tid] = __expf(rowm[tid] - mnew);
      rowm[tid]     = mnew;
    }
    __syncthreads();

    // P = exp(S - m_new) computed by all 256 threads in registers.
#pragma unroll
    for (int r = 0; r < 8; ++r) {
      int row = r + 8 * hf;
      float p = __expf(sreg[r] - rowm[row]);
      Ps[row][wave * 16 + m16] = (_Float16)p;
      float sm = red16_sum(p);
      if (m16 == 0) psum[row][wave] = sm;
    }
    __syncthreads();   // Ps + psum ready

    if (tid < 16) {
      float add = 0.0f;
#pragma unroll
      for (int w = 0; w < 8; ++w) add += psum[tid][w];
      rowl[tid] = rowl[tid] * rowscale[tid] + add;
    }

    // ---- rescale accumulator, then O += P V (A from Ps, B from Vs) ----
#pragma unroll
    for (int r = 0; r < 8; ++r) o[r] *= rowscale[r + 8 * hf];
#pragma unroll
    for (int ks = 0; ks < 4; ++ks) {
      v16h a, bmat;
#pragma unroll
      for (int j = 0; j < 8; ++j) {
        int k0 = ks * 32 + wmma_a_k(j, lane);
        a[2 * j]     = Ps[m16][k0];
        a[2 * j + 1] = Ps[m16][k0 + 1];
      }
#pragma unroll
      for (int i = 0; i < 16; ++i)
        bmat[i] = Vs[ks * 32 + hf * 8 + (i & 7) + (i >> 3) * 16][wave * 16 + m16];
      o = __builtin_amdgcn_wmma_f32_16x16x32_f16(false, a, false, bmat,
                                                 (short)0, o, false, false);
    }
    __syncthreads();   // Ps/Vs fully consumed before next stripe rewrites them
  }

  // ---- 1/l scaling, Lorentz normalization, output ----
#pragma unroll
  for (int r = 0; r < 8; ++r) {
    int row = r + 8 * hf;
    Os[row][wave * 16 + m16] = o[r] / rowl[row];
  }
  __syncthreads();
  if (tid < 16) {
    float t0 = Os[tid][0];
    float ss = 0.0f;
    for (int j = 1; j < DD; ++j) { float v = Os[tid][j]; ss += v * v; }
    float lor = -t0 * t0 + ss;
    rowscale[tid] = 1.0f / sqrtf(fmaxf(fabsf(lor), 1e-8f));
  }
  __syncthreads();
  // Vectorized output: float4 per thread iteration.
  for (int i = tid; i < 16 * (DD / 4); i += 256) {
    int r = i >> 5, c4 = (i & 31) * 4;
    float4 vv;
    vv.x = Os[r][c4 + 0] * rowscale[r];
    vv.y = Os[r][c4 + 1] * rowscale[r];
    vv.z = Os[r][c4 + 2] * rowscale[r];
    vv.w = Os[r][c4 + 3] * rowscale[r];
    *(float4*)&out[((size_t)bidx * LL + qbase + r) * DD + c4] = vv;
  }
}

// ---------------------------------------------------------------------------
// Launch
// ---------------------------------------------------------------------------
extern "C" void kernel_launch(void* const* d_in, const int* in_sizes, int n_in,
                              void* d_out, int out_size, void* d_ws, size_t ws_size,
                              hipStream_t stream) {
  (void)in_sizes; (void)n_in; (void)out_size; (void)ws_size;

  const float* query = (const float*)d_in[0];
  const float* key   = (const float*)d_in[1];
  const float* value = (const float*)d_in[2];
  const unsigned char* mask = (const unsigned char*)d_in[3]; // bool array
  const float* Wq = (const float*)d_in[4];
  const float* bq = (const float*)d_in[5];
  const float* sq = (const float*)d_in[6];
  const float* Wk = (const float*)d_in[7];
  const float* bk = (const float*)d_in[8];
  const float* sk = (const float*)d_in[9];
  const float* Wv = (const float*)d_in[10];
  const float* bv = (const float*)d_in[11];
  const float* sv = (const float*)d_in[12];
  const float* attn_scale = (const float*)d_in[13];
  const float* attn_bias  = (const float*)d_in[14];
  float* out = (float*)d_out;

  // Workspace: f16 Q (time negated), K, V — 3 * 8*2048*128 * 2B = 12.6 MB
  _Float16* Qh = (_Float16*)d_ws;
  _Float16* Kh = Qh + (size_t)BB * LL * DD;
  _Float16* Vh = Kh + (size_t)BB * LL * DD;

  dim3 block(256);
  dim3 grid_proj(BB * LL / 16);   // 1024 row tiles
  lorentz_proj_f16_wmma<<<grid_proj, block, 0, stream>>>(query, Wq, bq, sq, Qh, 1);
  lorentz_proj_f16_wmma<<<grid_proj, block, 0, stream>>>(key,   Wk, bk, sk, Kh, 0);
  lorentz_proj_f16_wmma<<<grid_proj, block, 0, stream>>>(value, Wv, bv, sv, Vh, 0);

  dim3 grid_attn(BB * (LL / 16)); // 1024 query tiles
  lorentz_flash_attn_wmma<<<grid_attn, block, 0, stream>>>(Qh, Kh, Vh, mask,
                                                           attn_scale, attn_bias, out);
}
